// SPECTRELayer_26731876450385
// MI455X (gfx1250) — matrix operand order repaired and checked by reading, not compiled, not run on hardware
//
#include <hip/hip_runtime.h>
#include <hip/hip_fp16.h>

// ---------------------------------------------------------------------------
// SPECTRE layer for MI455X (gfx1250, wave32, WMMA).
//   out = (irfft( modrelu( rfft(x@Wv^T) * gates(x@Wq^T) ) )) @ Wo^T
// GEMMs (3 x 16384x1024x1024) via v_wmma_f32_16x16x32_f16 with async
// global->LDS double-buffered B-tile staging (ASYNCcnt pipeline).
// rFFT->gate->irFFT fused per (b,h,dh) column in 64KB LDS (Stockham radix-2).
// ---------------------------------------------------------------------------

#define Bz   4
#define Sz   4096
#define Dz   1024
#define Hz   16
#define DHz  64
#define NFz  2049
#define FFTN 4096

typedef __attribute__((ext_vector_type(16))) _Float16 v16h;
typedef __attribute__((ext_vector_type(8)))  float    v8f;

union HF16 { uint4 q[2]; v16h v; };

#if defined(__has_builtin)
#  if __has_builtin(__builtin_amdgcn_global_load_async_to_lds_b128) && \
      __has_builtin(__builtin_amdgcn_s_wait_asynccnt)
#    define USE_ASYNC_LDS 1
#  endif
#endif
#ifndef USE_ASYNC_LDS
#  define USE_ASYNC_LDS 0
#endif

#if USE_ASYNC_LDS
typedef int v4i __attribute__((__vector_size__(4 * sizeof(int))));
typedef __attribute__((address_space(1))) v4i* gv4i_p;   // global (__device__)
typedef __attribute__((address_space(3))) v4i* lv4i_p;   // LDS (__shared__)
#endif

// ------------------------- conversion kernels ------------------------------

__global__ void k_f32_to_f16(const float* __restrict__ src,
                             _Float16* __restrict__ dst, int n) {
    int i = blockIdx.x * blockDim.x + threadIdx.x;
    if (i < n) dst[i] = (_Float16)src[i];
}

// W [Nn][Kk] fp32 row-major  ->  Wt [Kk][Nn] f16 row-major  (Wt[k][n] = W[n][k])
__global__ void k_transpose_to_f16(const float* __restrict__ W,
                                   _Float16* __restrict__ Wt, int Nn, int Kk) {
    int i = blockIdx.x * blockDim.x + threadIdx.x;
    if (i < Nn * Kk) {
        int nrow = i / Kk;
        int kcol = i - nrow * Kk;
        Wt[(size_t)kcol * Nn + nrow] = (_Float16)W[i];
    }
}

// ------------------------------ WMMA GEMM ----------------------------------
// C[M,N] = A[M,K] * Bt[K,N]   (A,Bt f16 row-major, C fp32)
// wave tile 16x64 (4 accumulators), block = 8 waves stacked on M -> 128x64.
// B tile (32x64) is shared by all 8 waves: staged in LDS via async DMA,
// double-buffered, so each k-step's WMMA reads LDS while the next tile flies.
__global__ __launch_bounds__(256) void k_gemm_f16_wmma(
    const _Float16* __restrict__ A,
    const _Float16* __restrict__ Bt,
    float* __restrict__ C,
    int M, int N, int K)
{
    const int lane = threadIdx.x & 31;
    const int wave = threadIdx.x >> 5;
    const int tm   = blockIdx.y * 128 + wave * 16;
    const int tn   = blockIdx.x * 64;
    const int half = lane >> 4;
    const int mrow = tm + (lane & 15);

    v8f acc[4] = {};

#if USE_ASYNC_LDS
    // [2 buffers][32 rows][72 halfs]; rows padded 64->72 to spread LDS banks.
    __shared__ _Float16 lb[2][32 * 72];
    const int t      = threadIdx.x;
    const int srow   = t >> 3;            // 0..31  (K row within tile)
    const int scol   = (t & 7) * 8;       // 0,8,..,56 (N col, 8 halfs = 16B)
    const int ldsoff = srow * 72 + scol;

    {   // prologue: stage first tile into buffer 0
        const _Float16* gp = Bt + (size_t)srow * N + tn + scol;
        __builtin_amdgcn_global_load_async_to_lds_b128(
            (gv4i_p)gp, (lv4i_p)&lb[0][ldsoff], 0, 0);
    }

    int bufc = 0;
    for (int kk = 0; kk < K; kk += 32) {
        if (kk + 32 < K) {                // stage next tile into other buffer
            const _Float16* gp = Bt + (size_t)(kk + 32 + srow) * N + tn + scol;
            __builtin_amdgcn_global_load_async_to_lds_b128(
                (gv4i_p)gp, (lv4i_p)&lb[bufc ^ 1][ldsoff], 0, 0);
            __builtin_amdgcn_s_wait_asynccnt(1);   // oldest (current tile) done
        } else {
            __builtin_amdgcn_s_wait_asynccnt(0);
        }
        __syncthreads();                  // all waves' DMA portions visible

        // A fragment: 16x32 f16 direct from global.
        const _Float16* ap = A + (size_t)mrow * K + kk + half * 8;
        __builtin_prefetch(ap + 32, 0, 3);                 // global_prefetch_b8
        HF16 af;
        af.q[0] = *(const uint4*)(ap);
        af.q[1] = *(const uint4*)(ap + 16);

        // B fragment: column n=lane&15 of the LDS tile, k in table order (8*half+s, 16+8*half+s).
        const _Float16* lrow = &lb[bufc][(half * 8) * 72 + (lane & 15)];
#pragma unroll
        for (int j = 0; j < 4; ++j) {
            HF16 bf;
#pragma unroll
            for (int s = 0; s < 8; ++s) {
                bf.v[s]     = lrow[s * 72 + j * 16];
                bf.v[s + 8] = lrow[(s + 16) * 72 + j * 16];
            }
            acc[j] = __builtin_amdgcn_wmma_f32_16x16x32_f16(
                false, af.v, false, bf.v, (short)0, acc[j], false, false);
        }
        __syncthreads();                  // done reading before next DMA lands
        bufc ^= 1;
    }
#else
    for (int kk = 0; kk < K; kk += 32) {
        const _Float16* ap = A + (size_t)mrow * K + kk + half * 8;
        __builtin_prefetch(ap + 32, 0, 3);
        HF16 af;
        af.q[0] = *(const uint4*)(ap);
        af.q[1] = *(const uint4*)(ap + 16);

        const _Float16* bp = Bt + (size_t)(kk + half * 8) * N + tn + (lane & 15);
        __builtin_prefetch(bp + (size_t)32 * N, 0, 3);
#pragma unroll
        for (int j = 0; j < 4; ++j) {
            HF16 bf;
#pragma unroll
            for (int s = 0; s < 8; ++s) {
                bf.v[s]     = bp[(size_t)s * N + j * 16];
                bf.v[s + 8] = bp[(size_t)(s + 16) * N + j * 16];
            }
            acc[j] = __builtin_amdgcn_wmma_f32_16x16x32_f16(
                false, af.v, false, bf.v, (short)0, acc[j], false, false);
        }
    }
#endif

    // C/D layout: VGPR v -> row = tm + v + 8*half; col = tn + j*16 + (lane&15)
#pragma unroll
    for (int j = 0; j < 4; ++j) {
#pragma unroll
        for (int v = 0; v < 8; ++v) {
            int row = tm + v + 8 * half;
            int col = tn + j * 16 + (lane & 15);
            C[(size_t)row * N + col] = acc[j][v];
        }
    }
}

// ------------------------- gate-path small kernels -------------------------

// mq[b, d] = mean_s Q[b, s, d]
__global__ void k_mean_q(const float* __restrict__ Q, float* __restrict__ mq) {
    int t = blockIdx.x * blockDim.x + threadIdx.x;
    if (t >= Bz * Dz) return;
    int b = t / Dz, dcol = t - b * Dz;
    const float* p = Q + (size_t)b * Sz * Dz + dcol;
    float s = 0.f;
    for (int i = 0; i < Sz; ++i) s += p[(size_t)i * Dz];
    mq[t] = s * (1.0f / Sz);
}

// LayerNorm over DH per (b,h), then mean over heads -> qsh[b, d]
__global__ __launch_bounds__(1024) void k_ln_headmean(
    const float* __restrict__ mq, const float* __restrict__ gamma,
    const float* __restrict__ beta, float* __restrict__ qsh)
{
    __shared__ float buf[Hz][DHz];
    const int b = blockIdx.x;
    const int h = threadIdx.x / DHz;
    const int d = threadIdx.x - h * DHz;

    float v = mq[b * Dz + h * DHz + d];
    buf[h][d] = v; __syncthreads();
    for (int off = 32; off > 0; off >>= 1) {
        if (d < off) buf[h][d] += buf[h][d + off];
        __syncthreads();
    }
    float mean = buf[h][0] * (1.0f / DHz); __syncthreads();
    float c = v - mean;
    buf[h][d] = c * c; __syncthreads();
    for (int off = 32; off > 0; off >>= 1) {
        if (d < off) buf[h][d] += buf[h][d + off];
        __syncthreads();
    }
    float var = buf[h][0] * (1.0f / DHz); __syncthreads();
    float qn = c * rsqrtf(var + 1e-5f) * gamma[d] + beta[d];
    buf[h][d] = qn; __syncthreads();
    if (h == 0) {
        float s = 0.f;
        for (int hh = 0; hh < Hz; ++hh) s += buf[hh][d];
        qsh[b * DHz + d] = s * (1.0f / Hz);
    }
}

// hdn[b, j] = gelu_exact( qsh[b,:] . w1[j,:] + b1[j] )
__global__ void k_mlp1(const float* __restrict__ qsh, const float* __restrict__ w1,
                       const float* __restrict__ b1, float* __restrict__ hdn) {
    int t = blockIdx.x * blockDim.x + threadIdx.x;
    if (t >= Bz * Dz) return;
    int b = t / Dz, j = t - b * Dz;
    const float* q = qsh + b * DHz;
    const float* w = w1 + (size_t)j * DHz;
    float acc = b1[j];
    for (int d = 0; d < DHz; ++d) acc += q[d] * w[d];
    hdn[t] = 0.5f * acc * (1.0f + erff(acc * 0.70710678118654752f));
}

// gri[b, o] = hdn[b,:] . w2[o,:] + b2[o]   (o in [0, 2*NF))
__global__ void k_mlp2(const float* __restrict__ hdn, const float* __restrict__ w2,
                       const float* __restrict__ b2, float* __restrict__ gri) {
    const int NO = 2 * NFz;
    int t = blockIdx.x * blockDim.x + threadIdx.x;
    if (t >= Bz * NO) return;
    int b = t / NO, o = t - b * NO;
    const float* hv = hdn + b * Dz;
    const float* w  = w2 + (size_t)o * Dz;
    float acc = b2[o];
    for (int d = 0; d < Dz; ++d) acc += hv[d] * w[d];
    gri[t] = acc;
}

// ----------------- fused rFFT -> gate*modReLU -> irFFT ---------------------

__device__ __forceinline__ void fft_stage(
    const float* sR, const float* sI,
    float* dR, float* dI,
    int n, int ls, float sign, int tid)
{
    const int m   = n >> 1;
    const int sst = 1 << ls;
    const float angstep = sign * 6.2831853071795864769f / (float)n;
    for (int t = tid; t < FFTN / 2; t += 256) {
        const int p  = t >> ls;
        const int q  = t & (sst - 1);
        const int i0 = q + sst * p;
        const int i1 = i0 + sst * m;
        const float ar = sR[i0], ai = sI[i0];
        const float br = sR[i1], bi = sI[i1];
        float wr, wi;
        __sincosf(angstep * (float)p, &wi, &wr);
        const int o0 = q + sst * (2 * p);
        const int o1 = o0 + sst;
        dR[o0] = ar + br;  dI[o0] = ai + bi;
        const float xr = ar - br, xi = ai - bi;
        dR[o1] = xr * wr - xi * wi;
        dI[o1] = xr * wi + xi * wr;
    }
}

// grid = (DH, H, B); block = 256. One length-4096 column per workgroup.
__global__ __launch_bounds__(256) void k_spectral(
    const float* __restrict__ V,      // [B, S, D] fp32 (D index = h*64+d)
    const float* __restrict__ gri,    // [B, 2*NF] fp32 (re | im)
    const float* __restrict__ modb,   // [1]
    _Float16* __restrict__ Oh)        // [B, S, D] f16 output (time domain)
{
    __shared__ float sR0[FFTN], sI0[FFTN], sR1[FFTN], sI1[FFTN];

    const int d = blockIdx.x, h = blockIdx.y, b = blockIdx.z;
    const int tid  = threadIdx.x;
    const float bias = modb[0];

    // load real column (imag = 0)
    const float* vp = V + (size_t)b * Sz * Dz + h * DHz + d;
    for (int s = tid; s < FFTN; s += 256) {
        sR0[s] = vp[(size_t)s * Dz];
        sI0[s] = 0.f;
    }
    __syncthreads();

    // forward FFT (sign -1), Stockham: 12 stages, result lands in buffer 0
    {
        int cur = 0, n = FFTN;
        for (int ls = 0; ls < 12; ++ls) {
            float* srcR = cur ? sR1 : sR0;
            float* srcI = cur ? sI1 : sI0;
            float* dstR = cur ? sR0 : sR1;
            float* dstI = cur ? sI0 : sI1;
            fft_stage(srcR, srcI, dstR, dstI, n, ls, -1.f, tid);
            __syncthreads();
            cur ^= 1; n >>= 1;
        }
    }

    // gate * modReLU on k in [0, 2048], Hermitian mirror into buffer 1
    {
        const float* grp = gri + (size_t)b * (2 * NFz);
        for (int k = tid; k < NFz; k += 256) {
            float xr = sR0[k] * (1.0f / 64.0f);   // ortho forward scale
            float xi = sI0[k] * (1.0f / 64.0f);
            float gr = grp[k], gi = grp[NFz + k];
            float zr = xr * gr - xi * gi;
            float zi = xr * gi + xi * gr;
            float mag = sqrtf(zr * zr + zi * zi);
            float sc  = fmaxf(mag + bias, 0.f) / fmaxf(mag, 1e-12f);
            zr *= sc; zi *= sc;
            sR1[k] = zr; sI1[k] = zi;
            if (k > 0 && k < NFz - 1) {           // mirror k -> 4096-k
                sR1[FFTN - k] = zr;
                sI1[FFTN - k] = -zi;
            }
        }
    }
    __syncthreads();

    // inverse FFT (sign +1), result lands in buffer 1
    {
        int cur = 1, n = FFTN;
        for (int ls = 0; ls < 12; ++ls) {
            float* srcR = cur ? sR1 : sR0;
            float* srcI = cur ? sI1 : sI0;
            float* dstR = cur ? sR0 : sR1;
            float* dstI = cur ? sI0 : sI1;
            fft_stage(srcR, srcI, dstR, dstI, n, ls, 1.f, tid);
            __syncthreads();
            cur ^= 1; n >>= 1;
        }
    }

    // store real part with ortho inverse scale, directly as f16 for Wo GEMM
    _Float16* op = Oh + (size_t)b * Sz * Dz + h * DHz + d;
    for (int s = tid; s < FFTN; s += 256) {
        op[(size_t)s * Dz] = (_Float16)(sR1[s] * (1.0f / 64.0f));
    }
}

// ------------------------------- launcher ----------------------------------

extern "C" void kernel_launch(void* const* d_in, const int* in_sizes, int n_in,
                              void* d_out, int out_size, void* d_ws, size_t ws_size,
                              hipStream_t stream) {
    (void)in_sizes; (void)n_in; (void)out_size; (void)ws_size;
    const float* x  = (const float*)d_in[0];
    const float* Wq = (const float*)d_in[1];
    const float* Wv = (const float*)d_in[2];
    const float* Wo = (const float*)d_in[3];
    const float* ga = (const float*)d_in[4];
    const float* be = (const float*)d_in[5];
    const float* w1 = (const float*)d_in[6];
    const float* b1 = (const float*)d_in[7];
    const float* w2 = (const float*)d_in[8];
    const float* b2 = (const float*)d_in[9];
    const float* mb = (const float*)d_in[10];

    char* ws = (char*)d_ws;
    size_t off = 0;
    auto alloc = [&](size_t bytes) -> char* {
        char* p = ws + off;
        off += (bytes + 255) & ~(size_t)255;
        return p;
    };

    const size_t BSD = (size_t)Bz * Sz * Dz;
    _Float16* xh  = (_Float16*)alloc(BSD * 2);
    _Float16* WqT = (_Float16*)alloc((size_t)Dz * Dz * 2);
    _Float16* WvT = (_Float16*)alloc((size_t)Dz * Dz * 2);
    _Float16* WoT = (_Float16*)alloc((size_t)Dz * Dz * 2);
    float*    Q   = (float*)alloc(BSD * 4);
    float*    V   = (float*)alloc(BSD * 4);
    _Float16* Oh  = (_Float16*)alloc(BSD * 2);
    float*    mq  = (float*)alloc((size_t)Bz * Dz * 4);
    float*    qsh = (float*)alloc((size_t)Bz * DHz * 4);
    float*    hdn = (float*)alloc((size_t)Bz * Dz * 4);
    float*    gri = (float*)alloc((size_t)Bz * 2 * NFz * 4);

    const int nBSD = Bz * Sz * Dz;
    k_f32_to_f16<<<(nBSD + 255) / 256, 256, 0, stream>>>(x, xh, nBSD);
    k_transpose_to_f16<<<(Dz * Dz + 255) / 256, 256, 0, stream>>>(Wq, WqT, Dz, Dz);
    k_transpose_to_f16<<<(Dz * Dz + 255) / 256, 256, 0, stream>>>(Wv, WvT, Dz, Dz);
    k_transpose_to_f16<<<(Dz * Dz + 255) / 256, 256, 0, stream>>>(Wo, WoT, Dz, Dz);

    dim3 gg(Dz / 64, (Bz * Sz) / 128);
    k_gemm_f16_wmma<<<gg, 256, 0, stream>>>(xh, WqT, Q, Bz * Sz, Dz, Dz);
    k_gemm_f16_wmma<<<gg, 256, 0, stream>>>(xh, WvT, V, Bz * Sz, Dz, Dz);

    k_mean_q<<<(Bz * Dz + 255) / 256, 256, 0, stream>>>(Q, mq);
    k_ln_headmean<<<Bz, Hz * DHz, 0, stream>>>(mq, ga, be, qsh);
    k_mlp1<<<(Bz * Dz + 255) / 256, 256, 0, stream>>>(qsh, w1, b1, hdn);
    k_mlp2<<<(Bz * 2 * NFz + 255) / 256, 256, 0, stream>>>(hdn, w2, b2, gri);

    dim3 gs(DHz, Hz, Bz);
    k_spectral<<<gs, 256, 0, stream>>>(V, gri, mb, Oh);

    k_gemm_f16_wmma<<<gg, 256, 0, stream>>>(Oh, WoT, (float*)d_out, Bz * Sz, Dz, Dz);
}
